// MHBAWithMask_19842748907883
// MI455X (gfx1250) — matrix operand order repaired and checked by reading, not compiled, not run on hardware
//
#include <hip/hip_runtime.h>
#include <hip/hip_bf16.h>

// ---- problem constants (match reference) ----
#define BB 4
#define LL 1024
#define EE 1024
#define HH 16
#define DD 64
#define BH (BB * HH)          // 64
#define HEAD_ELEMS (LL * DD)  // 65536 per (b,h)

typedef _Float16 v16h __attribute__((ext_vector_type(16)));
typedef float    v8f  __attribute__((ext_vector_type(8)));

union AFrag { v16h v; _Float16 h[16]; uint4 q[2]; };   // 32B: A or B matrix frag (16 halfs/lane)
union HFrag8 { uint4 q; _Float16 h[8]; };

// reduce across the 16 lanes of a half-wave group (xor offsets < 16 stay in group)
__device__ __forceinline__ float rowmax16(float v) {
#pragma unroll
  for (int off = 8; off > 0; off >>= 1) v = fmaxf(v, __shfl_xor(v, off, 32));
  return v;
}
__device__ __forceinline__ float rowsum16(float v) {
#pragma unroll
  for (int off = 8; off > 0; off >>= 1) v += __shfl_xor(v, off, 32);
  return v;
}

// CDNA5 async global->LDS copy (VGLOBAL encoding, GV mode):
//   GLOBAL_LOAD_ASYNC_TO_LDS_B128 vdst(LDS byte addr), vaddr(64b global), off
// Tracked by ASYNCcnt; drain with s_wait_asynccnt before the block barrier.
__device__ __forceinline__ void async_load_lds_b128(unsigned lds_off, const void* gaddr) {
  asm volatile("global_load_async_to_lds_b128 %0, %1, off"
               :: "v"(lds_off), "v"(gaddr) : "memory");
}
__device__ __forceinline__ void wait_async0() {
  asm volatile("s_wait_asynccnt 0x0" ::: "memory");
}
// flat aperture rule: low 32 bits of the flat address -> wave-relative LDS byte address
__device__ __forceinline__ unsigned lds_addr_of(const void* p) {
  return (unsigned)(unsigned long long)p;
}

// -------------------------------------------------------------------------
// Kernel 1: keys -> bernoulli mask -> softmax over SEQ dim (per b,h,d column)
// output f16 [B,H,L,D] row-major (l major, d minor)
// grid: BH blocks x 64 threads (thread = one d column)
// -------------------------------------------------------------------------
__global__ void ksoft_kernel(const float* __restrict__ keys,
                             const int* __restrict__ bern,
                             _Float16* __restrict__ ksoft) {
  const int bh = blockIdx.x, b = bh >> 4, h = bh & 15, d = threadIdx.x;
  const int* bm = bern + (size_t)bh * LL;
  const float* kcol = keys + (size_t)b * LL * EE + h * DD + d;
  float mx = -INFINITY;
  for (int l = 0; l < LL; ++l)
    if (bm[l]) mx = fmaxf(mx, kcol[(size_t)l * EE]);
  _Float16* kout = ksoft + (size_t)bh * HEAD_ELEMS + d;
  float s = 0.f;
  for (int l = 0; l < LL; ++l) {
    float p = bm[l] ? __expf(kcol[(size_t)l * EE] - mx) : 0.f;
    s += p;
    kout[l * DD] = (_Float16)p;
  }
  const float inv = (s > 0.f) ? 1.f / s : 0.f;
  for (int l = 0; l < LL; ++l)
    kout[l * DD] = (_Float16)((float)kout[l * DD] * inv);
}

// -------------------------------------------------------------------------
// Kernel 2: grouped 3x3 conv over (L,D) per head channel, + bias -> f16 [B,H,L,D]
// one thread per output element
// -------------------------------------------------------------------------
__global__ void qconv_kernel(const float* __restrict__ query,
                             const float* __restrict__ cw,
                             const float* __restrict__ cb,
                             _Float16* __restrict__ qc) {
  const int g = blockIdx.x * 256 + threadIdx.x;   // B*H*L*D = 4M
  const int d = g & 63, l = (g >> 6) & (LL - 1), bh = g >> 16;
  const int h = bh & 15, b = bh >> 4;
  const float* wp = cw + h * 9;
  float acc = cb[h];
#pragma unroll
  for (int i = 0; i < 3; ++i) {
    const int ll = l + i - 1;
    if (ll < 0 || ll >= LL) continue;
    const float* qrow = query + ((size_t)(b * LL + ll)) * EE + h * DD;
#pragma unroll
    for (int j = 0; j < 3; ++j) {
      const int dd = d + j - 1;
      if (dd >= 0 && dd < DD) acc += qrow[dd] * wp[i * 3 + j];
    }
  }
  qc[(size_t)bh * HEAD_ELEMS + l * DD + d] = (_Float16)acc;
}

// -------------------------------------------------------------------------
// Kernel 3: vproj[b,h,l,e] = sum_d values[b,l,h*64+d] * Wv[e,d]   (WMMA)
// grid: (L/64, BH), block 128 (4 waves; each wave = 16 rows x full 64 cols)
// -------------------------------------------------------------------------
__global__ __launch_bounds__(128) void vproj_kernel(const float* __restrict__ values,
                                                    const float* __restrict__ Wv,
                                                    _Float16* __restrict__ vproj) {
  __shared__ __align__(16) _Float16 sW[DD * DD];   // Wv as f16, row-major [e][d]
  const int bh = blockIdx.y, b = bh >> 4, h = bh & 15;
  for (int i = threadIdx.x; i < DD * DD; i += 128) sW[i] = (_Float16)Wv[i];
  __syncthreads();

  const int w = threadIdx.x >> 5, lane = threadIdx.x & 31;
  const int hi = lane >> 4, ln = lane & 15;
  const int l0 = (blockIdx.x * 4 + w) * 16;

  // A-frags (16x32 f16 layout): lane row = ln; half j -> d = s*32 + hi*16 + j (contiguous runs)
  AFrag a0, a1;
  const float* vrow = values + ((size_t)(b * LL + l0 + ln)) * EE + h * DD;
#pragma unroll
  for (int j = 0; j < 16; ++j) {
    const int d0 = hi * 16 + j;
    a0.h[j] = (_Float16)vrow[d0];
    a1.h[j] = (_Float16)vrow[32 + d0];
  }

#pragma unroll
  for (int n = 0; n < 4; ++n) {
    const int e = n * 16 + ln;
    // B-frags (32x16): lane col = ln, half j -> kd = s*32 + hi*16 + j (contiguous runs)
    AFrag b0, b1;
    const _Float16* wr = sW + e * DD + hi * 16;
    b0.q[0] = *(const uint4*)(wr);      b0.q[1] = *(const uint4*)(wr + 8);
    b1.q[0] = *(const uint4*)(wr + 32); b1.q[1] = *(const uint4*)(wr + 40);
    v8f c = {};
    c = __builtin_amdgcn_wmma_f32_16x16x32_f16(false, a0.v, false, b0.v, (short)0, c, false, false);
    c = __builtin_amdgcn_wmma_f32_16x16x32_f16(false, a1.v, false, b1.v, (short)0, c, false, false);
    _Float16* orow = vproj + (size_t)bh * HEAD_ELEMS + (size_t)l0 * DD + e;
#pragma unroll
    for (int i = 0; i < 8; ++i)
      orow[(i + 8 * hi) * DD] = (_Float16)c[i];
  }
}

// -------------------------------------------------------------------------
// Kernel 4: fused flash attention
//   energy = qc . ksoft^T, mask(pad&causal -> -1e20), online softmax (scale 1/32),
//   out += attn . vproj ; grid: (L/128, BH), block 256 = 8 waves x 16 q-rows
// -------------------------------------------------------------------------
__global__ __launch_bounds__(256) void attn_kernel(const _Float16* __restrict__ qc,
                                                   const _Float16* __restrict__ ksoft,
                                                   const _Float16* __restrict__ vproj,
                                                   const int* __restrict__ pad,
                                                   const int* __restrict__ caus,
                                                   float* __restrict__ out) {
  __shared__ __align__(16) _Float16 sK[32 * DD];        // K chunk, row-major [kk][d]
  __shared__ __align__(16) _Float16 sVt[DD * 32];       // V chunk transposed [d][kk]
  __shared__ __align__(16) _Float16 sP[8][16 * 32];     // per-wave P tiles
  __shared__ unsigned char sMask[LL];

  const int bh = blockIdx.y, b = bh >> 4, h = bh & 15;
  const int w = threadIdx.x >> 5, lane = threadIdx.x & 31;
  const int hi = lane >> 4, ln = lane & 15;
  const int q0 = blockIdx.x * 128 + w * 16;

  const _Float16* qcp = qc    + (size_t)bh * HEAD_ELEMS;
  const _Float16* kp  = ksoft + (size_t)bh * HEAD_ELEMS;
  const _Float16* vp  = vproj + (size_t)bh * HEAD_ELEMS;

  // combined mask for this batch
  for (int i = threadIdx.x; i < LL; i += 256)
    sMask[i] = (pad[b * LL + i] != 0) && (caus[b * LL + i] != 0);

  // Q tile A-frags (held for the whole loop)
  AFrag aLo, aHi;
  {
    const _Float16* qrow = qcp + (size_t)(q0 + ln) * DD;
    aLo.q[0] = *(const uint4*)(qrow + hi * 16);
    aLo.q[1] = *(const uint4*)(qrow + 8 + hi * 16);
    aHi.q[0] = *(const uint4*)(qrow + 32 + hi * 16);
    aHi.q[1] = *(const uint4*)(qrow + 40 + hi * 16);
  }

  const unsigned sK_lds = lds_addr_of(&sK[0]) + threadIdx.x * 16u;

  float m[8], l[8];
  v8f acc[4];
#pragma unroll
  for (int i = 0; i < 8; ++i) { m[i] = -INFINITY; l[i] = 0.f; }
#pragma unroll
  for (int n = 0; n < 4; ++n) { v8f z = {}; acc[n] = z; }

  for (int kc = 0; kc < LL / 32; ++kc) {
    const int k0 = kc * 32;
    __syncthreads();
    // stage K chunk via async DMA: 256 lanes x 16B = 4KB contiguous (no VGPR traffic)
    async_load_lds_b128(sK_lds, kp + (size_t)k0 * DD + threadIdx.x * 8);
    // stage V chunk transposed: thread -> (kk, 8-wide d segment)
    {
      const int kk = threadIdx.x >> 3, ds_ = (threadIdx.x & 7) * 8;
      HFrag8 t; t.q = *(const uint4*)(vp + (size_t)(k0 + kk) * DD + ds_);
#pragma unroll
      for (int j = 0; j < 8; ++j) sVt[(ds_ + j) * 32 + kk] = t.h[j];
    }
    if (kc + 1 < LL / 32) {
      __builtin_prefetch(kp + (size_t)(k0 + 32) * DD + (threadIdx.x << 3), 0, 1);
      __builtin_prefetch(vp + (size_t)(k0 + 32) * DD + (threadIdx.x << 3), 0, 1);
    }
    wait_async0();        // drain this wave's ASYNCcnt before signaling the barrier
    __syncthreads();

    // ---- energy: two 16x16 C tiles (32 k columns) ----
    v8f c[2];
#pragma unroll
    for (int ns = 0; ns < 2; ++ns) {
      AFrag bk0, bk1;   // B 32x16: lane col = k-row ns*16+ln, halfs = contiguous d run
      const _Float16* krow = sK + (ns * 16 + ln) * DD + hi * 16;
      bk0.q[0] = *(const uint4*)(krow);      bk0.q[1] = *(const uint4*)(krow + 8);
      bk1.q[0] = *(const uint4*)(krow + 32); bk1.q[1] = *(const uint4*)(krow + 40);
      v8f cz = {};
      cz = __builtin_amdgcn_wmma_f32_16x16x32_f16(false, aLo.v, false, bk0.v, (short)0, cz, false, false);
      cz = __builtin_amdgcn_wmma_f32_16x16x32_f16(false, aHi.v, false, bk1.v, (short)0, cz, false, false);
      c[ns] = cz;
    }

    // scale + mask (mask is per k-column == per lane)
#pragma unroll
    for (int ns = 0; ns < 2; ++ns) {
      const bool bad = sMask[k0 + ns * 16 + ln] != 0;
#pragma unroll
      for (int i = 0; i < 8; ++i) {
        const float e = c[ns][i] * 0.03125f;     // 1/sqrt(E)
        c[ns][i] = bad ? -1e20f : e;
      }
    }

    // ---- online softmax (rows live in vgpr slot, columns across 16 lanes) ----
    float p0[8], p1[8];
#pragma unroll
    for (int i = 0; i < 8; ++i) {
      const float rmax = rowmax16(fmaxf(c[0][i], c[1][i]));
      const float mn   = fmaxf(m[i], rmax);
      const float alpha = __expf(m[i] - mn);
      p0[i] = __expf(c[0][i] - mn);
      p1[i] = __expf(c[1][i] - mn);
      const float rs = rowsum16(p0[i] + p1[i]);
      l[i] = l[i] * alpha + rs;
      m[i] = mn;
#pragma unroll
      for (int n = 0; n < 4; ++n) acc[n][i] *= alpha;
    }

    // ---- bounce P through LDS to re-layout C(16x32) -> A(16x32) ----
    _Float16* pw = &sP[w][0];
#pragma unroll
    for (int i = 0; i < 8; ++i) {
      pw[(i + 8 * hi) * 32 + ln]      = (_Float16)p0[i];
      pw[(i + 8 * hi) * 32 + 16 + ln] = (_Float16)p1[i];
    }
    asm volatile("s_wait_dscnt 0" ::: "memory");   // wave-local LDS RAW fence
    AFrag pa;
    const _Float16* prow = pw + ln * 32 + hi * 16;
    pa.q[0] = *(const uint4*)(prow);
    pa.q[1] = *(const uint4*)(prow + 8);

    // ---- O += P(16x32) . V(32x64): 4 WMMAs over d subtiles ----
#pragma unroll
    for (int n = 0; n < 4; ++n) {
      AFrag bv;   // B 32x16: lane col = d, halfs = contiguous kk run (V transposed)
      const _Float16* vrow = sVt + (n * 16 + ln) * 32 + hi * 16;
      bv.q[0] = *(const uint4*)(vrow);
      bv.q[1] = *(const uint4*)(vrow + 8);
      acc[n] = __builtin_amdgcn_wmma_f32_16x16x32_f16(false, pa.v, false, bv.v, (short)0, acc[n], false, false);
    }
  }

  // ---- epilogue: normalize and write [b, q, h*64 + d] ----
#pragma unroll
  for (int i = 0; i < 8; ++i) {
    const int qg = q0 + i + 8 * hi;
    const float inv = 1.0f / l[i];
    float* orow = out + ((size_t)(b * LL + qg)) * EE + h * DD;
#pragma unroll
    for (int n = 0; n < 4; ++n) orow[n * 16 + ln] = acc[n][i] * inv;
  }
}

// -------------------------------------------------------------------------
extern "C" void kernel_launch(void* const* d_in, const int* in_sizes, int n_in,
                              void* d_out, int out_size, void* d_ws, size_t ws_size,
                              hipStream_t stream) {
  const float* query  = (const float*)d_in[0];
  const float* keys   = (const float*)d_in[1];
  const float* values = (const float*)d_in[2];
  const int*   pad    = (const int*)d_in[3];
  const int*   caus   = (const int*)d_in[4];
  const int*   bern   = (const int*)d_in[5];
  const float* cw     = (const float*)d_in[6];
  const float* cb     = (const float*)d_in[7];
  const float* Wv     = (const float*)d_in[8];
  float* out = (float*)d_out;

  const size_t headTotal = (size_t)BH * HEAD_ELEMS;  // 4M f16 elems each
  _Float16* ksoft = (_Float16*)d_ws;
  _Float16* vproj = ksoft + headTotal;
  _Float16* qc    = vproj + headTotal;

  ksoft_kernel<<<dim3(BH), dim3(64), 0, stream>>>(keys, bern, ksoft);
  qconv_kernel<<<dim3((BB * HH * LL * DD) / 256), dim3(256), 0, stream>>>(query, cw, cb, qc);
  vproj_kernel<<<dim3(LL / 64, BH), dim3(128), 0, stream>>>(values, Wv, vproj);
  attn_kernel<<<dim3(LL / 128, BH), dim3(256), 0, stream>>>(qc, ksoft, vproj, pad, caus, out);
}
